// SparseMax_VerticalSAMS_6425271074850
// MI455X (gfx1250) — matrix operand order repaired and checked by reading, not compiled, not run on hardware
//
#include <hip/hip_runtime.h>
#include <hip/hip_bf16.h>
#include <stdint.h>

// ---------------------------------------------------------------------------
// SparseMax VerticalSAMS (MoE) for gfx1250 / MI455X.
// - GEMMs on v_wmma_f32_16x16x32_f16 (f32 accumulate)
// - Expert/gate weights staged block-wide into LDS via gfx1250 async copies
//   (global_load_async_to_lds_b128 + s_wait_asynccnt), read back as ds_load.
// ---------------------------------------------------------------------------

#define USE_ASYNC_LDS 1

typedef __attribute__((ext_vector_type(16))) _Float16 v16h;
typedef __attribute__((ext_vector_type(8)))  float    v8f;

#define F_FIELDS   10
#define EMB_D      16
#define IN_DIM     160      // F * D
#define HG         256
#define KEXP       16
#define HE         256
#define KT1        5        // IN_DIM / 32 k-tiles
#define NT1        16       // 256 / 16 n-tiles
#define HALF_NT    8
// One 16x32 B fragment = 32 lanes * 8 dwords = 256 dwords (1 KB).
#define FRAG_DW    256
#define STAGE_DW   (KT1 * HALF_NT * FRAG_DW)   // 10240 dwords = 40 KB
#define STAGE_B128 (STAGE_DW / 4)              // 2560 16-byte units

// ---------------- fragment loaders ----------------

// A fragment: 16x32 f16 tile from row-major [M, ld].
// lanes 0-15 / 16-31: same 16 rows; half selects K+{0,8} and K+{16,24} groups.
__device__ inline v16h load_a_frag(const _Float16* __restrict__ A,
                                   int m0, int k0, int ld) {
    int lane = threadIdx.x & 31;
    int m    = m0 + (lane & 15);
    int half = lane >> 4;
    const _Float16* row = A + (size_t)m * ld;
    v16h a;
#pragma unroll
    for (int i = 0; i < 4; ++i) {
        int k      = k0 + half * 8 + 2 * i;
        a[2*i]     = row[k];
        a[2*i + 1] = row[k + 1];
    }
#pragma unroll
    for (int i = 0; i < 4; ++i) {
        int k          = k0 + 16 + half * 8 + 2 * i;
        a[8 + 2*i]     = row[k];
        a[8 + 2*i + 1] = row[k + 1];
    }
    return a;
}

// B fragment straight from pre-swizzled global weights (used by gate layer-2).
__device__ inline v16h load_b_frag(const uint32_t* __restrict__ Bp,
                                   int kt, int nt, int NT) {
    int lane = threadIdx.x & 31;
    const uint32_t* p = Bp + ((size_t)(kt * NT + nt) * 32 + lane) * 8;
    return *reinterpret_cast<const v16h*>(p);
}

// Stage 40 KB of swizzled weights (kt=0..4, ntl=0..7 of half hh) into LDS.
// gfx1250 async copy path: ASYNCcnt-tracked, no VGPR data movement.
__device__ inline void stage_weights_half(uint32_t* sh,
                                          const uint32_t* __restrict__ Bp,
                                          int hh) {
    for (int tt = threadIdx.x; tt < STAGE_B128; tt += 256) {
        int f   = tt >> 6;           // fragment 0..39
        int d   = tt & 63;           // 16B unit within fragment
        int kt  = f >> 3;
        int ntl = f & 7;
        const uint32_t* g = Bp + ((size_t)(kt * NT1 + hh * HALF_NT + ntl) * FRAG_DW
                                  + d * 4);
#if USE_ASYNC_LDS
        // LDS aperture maps addr[31:0] -> LDS offset, so the low 32 bits of a
        // generic pointer to __shared__ are exactly the async VDST operand.
        uint32_t ldsoff = (uint32_t)(uintptr_t)(sh + tt * 4);
        uint64_t gaddr  = (uint64_t)(uintptr_t)g;
        asm volatile("global_load_async_to_lds_b128 %0, %1, off"
                     :: "v"(ldsoff), "v"(gaddr)
                     : "memory");
#else
        *(uint4*)(sh + tt * 4) = *(const uint4*)g;
#endif
    }
#if USE_ASYNC_LDS
    asm volatile("s_wait_asynccnt 0x0" ::: "memory");
#endif
}

// ---------------- kernel 1: embedding gather + f32->f16 ----------------

__global__ void gather_convert_f16(const float* __restrict__ emb,
                                   const int* __restrict__ idx,
                                   _Float16* __restrict__ out,
                                   int B) {
    int total = B * IN_DIM;
    for (int t = blockIdx.x * blockDim.x + threadIdx.x; t < total;
         t += gridDim.x * blockDim.x) {
        int d  = t % EMB_D;
        int f  = (t / EMB_D) % F_FIELDS;
        int b  = t / IN_DIM;
        int id = idx[b * F_FIELDS + f];
        out[t] = (_Float16)emb[(size_t)id * EMB_D + d];
    }
}

// ---------------- kernel 2: weight swizzle (f32 -> WMMA B-frag f16) --------

__global__ void swizzle_weights(const float* __restrict__ W,  // [K, N] row major
                                uint32_t* __restrict__ Bp,
                                int Kdim, int N) {
    int Ktiles = Kdim >> 5;
    int NT     = N >> 4;
    int total  = Ktiles * NT * 32 * 8;
    for (int t = blockIdx.x * blockDim.x + threadIdx.x; t < total;
         t += gridDim.x * blockDim.x) {
        int j    = t & 7;
        int lane = (t >> 3) & 31;
        int nt   = (t >> 8) % NT;
        int kt   = (t >> 8) / NT;
        int n    = nt * 16 + (lane & 15);
        int k    = kt * 32 + (lane >> 4) * 8 + (j >> 2) * 8 + 2 * j;
        union { _Float16 h[2]; uint32_t u; } p;
        p.h[0] = (_Float16)W[(size_t)k * N + n];
        p.h[1] = (_Float16)W[(size_t)(k + 1) * N + n];
        Bp[t] = p.u;
    }
}

// ---------------- kernel 3: gate layer-1 GEMM + relu (f16 out) ------------
// Block = 8 waves = 8 row-tiles; W1 (80KB) staged to LDS in two 40KB halves.

__global__ __launch_bounds__(256) void gate_mlp1_wmma(
        const _Float16* __restrict__ A,     // [B,160]
        const uint32_t* __restrict__ W1p,
        const float* __restrict__ bias,     // [256]
        _Float16* __restrict__ H,           // [B,256]
        int M) {
    __shared__ __align__(16) uint32_t wsh[STAGE_DW];
    int wave = threadIdx.x >> 5;
    int lane = threadIdx.x & 31;
    int half = lane >> 4;
    int mt   = blockIdx.x * 8 + wave;

    v16h afr[KT1];
#pragma unroll
    for (int kt = 0; kt < KT1; ++kt)
        afr[kt] = load_a_frag(A, mt * 16, kt * 32, IN_DIM);

    for (int hh = 0; hh < 2; ++hh) {
        __syncthreads();
        stage_weights_half(wsh, W1p, hh);
        __syncthreads();
#pragma unroll
        for (int ntl = 0; ntl < HALF_NT; ++ntl) {
            v8f acc = {};
#pragma unroll
            for (int kt = 0; kt < KT1; ++kt) {
                const uint32_t* p = wsh + ((kt * HALF_NT + ntl) * 32 + lane) * 8;
                v16h b = *reinterpret_cast<const v16h*>(p);
                acc = __builtin_amdgcn_wmma_f32_16x16x32_f16(
                          false, afr[kt], false, b, (short)0, acc, false, false);
            }
            int   nt = hh * HALF_NT + ntl;
            int   n  = nt * 16 + (lane & 15);
            float bn = bias[n];
#pragma unroll
            for (int r = 0; r < 8; ++r) {
                int   m = mt * 16 + half * 8 + r;
                float v = acc[r] + bn;
                v = v > 0.0f ? v : 0.0f;
                H[(size_t)m * HG + n] = (_Float16)v;
            }
        }
    }
}

// ---------------- kernel 4: gate layer-2 GEMM (f32 logits) ----------------

__global__ void gate_mlp2_wmma(const _Float16* __restrict__ H,   // [B,256]
                               const uint32_t* __restrict__ W2p,
                               const float* __restrict__ bias,   // [16]
                               float* __restrict__ logits,       // [B,16]
                               int M) {
    const int Ktiles = HG >> 5;
    int wave = (blockIdx.x * blockDim.x + threadIdx.x) >> 5;
    if (wave >= (M >> 4)) return;
    int mt = wave;

    v8f acc = {};
#pragma unroll
    for (int kt = 0; kt < Ktiles; ++kt) {
        v16h a = load_a_frag(H, mt * 16, kt * 32, HG);
        v16h b = load_b_frag(W2p, kt, 0, 1);
        acc = __builtin_amdgcn_wmma_f32_16x16x32_f16(false, a, false, b,
                                                     (short)0, acc, false, false);
    }
    int lane = threadIdx.x & 31;
    int n    = lane & 15;
    int half = lane >> 4;
    float bn = bias[n];
#pragma unroll
    for (int r = 0; r < 8; ++r) {
        int m = mt * 16 + half * 8 + r;
        logits[(size_t)m * KEXP + n] = acc[r] + bn;
    }
}

// ---------------- kernel 5: entmax-1.5 bisection + loss partials ----------

__global__ void entmax_bisect_kernel(const float* __restrict__ logits,
                                     float* __restrict__ gate,   // [B,16]
                                     float* __restrict__ accum,  // [17]
                                     int B) {
    __shared__ float sacc[17];
    if (threadIdx.x < 17) sacc[threadIdx.x] = 0.0f;
    __syncthreads();

    int b = blockIdx.x * blockDim.x + threadIdx.x;
    float g[KEXP];
    float spa_local = 0.0f;
    bool active = (b < B);
    if (active) {
        float Xs[KEXP];
        float mx = -3.4e38f;
#pragma unroll
        for (int k = 0; k < KEXP; ++k) {
            Xs[k] = logits[(size_t)b * KEXP + k] * 0.5f;   // (alpha-1) = 0.5
            mx = Xs[k] > mx ? Xs[k] : mx;
        }
        float tau_lo = mx - 1.0f;
        float tau_hi = mx - 0.25f;                         // (1/16)^0.5
        float f_lo = -1.0f;
#pragma unroll
        for (int k = 0; k < KEXP; ++k) {
            float d = Xs[k] - tau_lo;
            if (d > 0.0f) f_lo += d * d;                   // p = clip^2
        }
        float dm = tau_hi - tau_lo;
        float tau_m = tau_lo;
        for (int it = 0; it < 50; ++it) {
            dm *= 0.5f;
            tau_m = tau_lo + dm;
            float f_m = -1.0f;
#pragma unroll
            for (int k = 0; k < KEXP; ++k) {
                float d = Xs[k] - tau_m;
                if (d > 0.0f) f_m += d * d;
            }
            if (f_m * f_lo >= 0.0f) tau_lo = tau_m;
        }
        float s = 0.0f;
#pragma unroll
        for (int k = 0; k < KEXP; ++k) {
            float d = Xs[k] - tau_m;
            g[k] = d > 0.0f ? d * d : 0.0f;
            s += g[k];
        }
        float inv = 1.0f / s;
        float ss  = 0.0f;
#pragma unroll
        for (int k = 0; k < KEXP; ++k) {
            g[k] *= inv;
            gate[(size_t)b * KEXP + k] = g[k];
            ss += g[k] * g[k];
        }
        spa_local = sqrtf(ss);
    }

    if (active) {
#pragma unroll
        for (int k = 0; k < KEXP; ++k) atomicAdd(&sacc[k], g[k]);
        atomicAdd(&sacc[16], spa_local);
    }
    __syncthreads();
    if (threadIdx.x < 17) atomicAdd(&accum[threadIdx.x], sacc[threadIdx.x]);
}

// ---------------- kernel 6: fused expert MLP (WMMA + dot-reduce) ----------
// Block = 8 waves = 8 row-tiles of ONE expert; expert W1 staged via async LDS.

__global__ __launch_bounds__(256) void expert_fused_wmma(
        const _Float16* __restrict__ A,   // [B,160]
        const uint32_t* __restrict__ W1p, // [K][frag]
        const float* __restrict__ b1,     // [K,256]
        const float* __restrict__ w2,     // [K,256]
        const float* __restrict__ b2,     // [K]
        float* __restrict__ eo,           // [B,K]
        int M) {
    __shared__ __align__(16) uint32_t wsh[STAGE_DW];
    const int fragPerExpert = KT1 * NT1 * FRAG_DW;
    int ke  = blockIdx.x & (KEXP - 1);
    int mtg = blockIdx.x >> 4;
    int wave = threadIdx.x >> 5;
    int lane = threadIdx.x & 31;
    int half = lane >> 4;
    int mt   = mtg * 8 + wave;
    const uint32_t* Bp = W1p + (size_t)ke * fragPerExpert;

    // A fragments resident in VGPRs for the whole kernel.
    v16h afr[KT1];
#pragma unroll
    for (int kt = 0; kt < KT1; ++kt)
        afr[kt] = load_a_frag(A, mt * 16, kt * 32, IN_DIM);

    float part[8] = {0, 0, 0, 0, 0, 0, 0, 0};
    for (int hh = 0; hh < 2; ++hh) {
        __syncthreads();
        stage_weights_half(wsh, Bp, hh);
        __syncthreads();
#pragma unroll
        for (int ntl = 0; ntl < HALF_NT; ++ntl) {
            v8f acc = {};
#pragma unroll
            for (int kt = 0; kt < KT1; ++kt) {
                const uint32_t* p = wsh + ((kt * HALF_NT + ntl) * 32 + lane) * 8;
                v16h bfr = *reinterpret_cast<const v16h*>(p);
                acc = __builtin_amdgcn_wmma_f32_16x16x32_f16(
                          false, afr[kt], false, bfr, (short)0, acc, false, false);
            }
            int   n  = (hh * HALF_NT + ntl) * 16 + (lane & 15);
            float bn = b1[ke * HE + n];
            float wn = w2[ke * HE + n];
#pragma unroll
            for (int r = 0; r < 8; ++r) {
                float v = acc[r] + bn;
                v = v > 0.0f ? v : 0.0f;
                part[r] += v * wn;                 // expert layer-2 dot, fused
            }
        }
    }

    // reduce over the 16 n-lanes of each half (wave32 shuffles)
#pragma unroll
    for (int off = 1; off < 16; off <<= 1) {
#pragma unroll
        for (int r = 0; r < 8; ++r)
            part[r] += __shfl_xor(part[r], off, 32);
    }
    if ((lane & 15) == 0) {
        float bb = b2[ke];
#pragma unroll
        for (int r = 0; r < 8; ++r) {
            int m = mt * 16 + half * 8 + r;
            eo[(size_t)m * KEXP + ke] = part[r] + bb;
        }
    }
}

// ---------------- kernel 7: combine + losses ----------------

__global__ void finalize_kernel(const float* __restrict__ gate,
                                const float* __restrict__ eo,
                                const float* __restrict__ accum,
                                float* __restrict__ out,
                                int B) {
    int b = blockIdx.x * blockDim.x + threadIdx.x;
    if (b < B) {
        float y = 0.0f;
#pragma unroll
        for (int k = 0; k < KEXP; ++k)
            y += gate[(size_t)b * KEXP + k] * eo[(size_t)b * KEXP + k];
        out[b] = y;
    }
    if (b == 0) {
        float mean = 0.0f;
#pragma unroll
        for (int k = 0; k < KEXP; ++k) mean += accum[k];
        mean *= (1.0f / KEXP);
        float var = 0.0f;
#pragma unroll
        for (int k = 0; k < KEXP; ++k) {
            float d = accum[k] - mean;
            var += d * d;
        }
        var *= (1.0f / (KEXP - 1));                    // ddof=1
        out[B]     = var / (mean * mean + 1e-10f);     // imp_loss
        out[B + 1] = accum[16] / (float)B;             // spa_loss
    }
}

// ---------------------------------------------------------------------------

extern "C" void kernel_launch(void* const* d_in, const int* in_sizes, int n_in,
                              void* d_out, int out_size, void* d_ws, size_t ws_size,
                              hipStream_t stream) {
    const float* input_emb_w = (const float*)d_in[0];
    const float* sql_emb_w   = (const float*)d_in[1];
    const float* gate_w1     = (const float*)d_in[2];
    const float* gate_b1     = (const float*)d_in[3];
    const float* gate_w2     = (const float*)d_in[4];
    const float* gate_b2     = (const float*)d_in[5];
    const float* exp_w1      = (const float*)d_in[6];
    const float* exp_b1      = (const float*)d_in[7];
    const float* exp_w2      = (const float*)d_in[8];
    const float* exp_b2      = (const float*)d_in[9];
    const int*   x_idx       = (const int*)d_in[10];
    const int*   sql_idx     = (const int*)d_in[11];
    float* out = (float*)d_out;

    const int B = in_sizes[10] / F_FIELDS;   // 65536

    // ---- workspace layout ----
    char* ws = (char*)d_ws;
    size_t off = 0;
    auto take = [&](size_t bytes) {
        char* p = ws + off;
        off += (bytes + 255) & ~(size_t)255;
        return p;
    };
    _Float16* xf16    = (_Float16*)take((size_t)B * IN_DIM * 2);
    _Float16* sql16   = (_Float16*)take((size_t)B * IN_DIM * 2);
    _Float16* h16     = (_Float16*)take((size_t)B * HG * 2);
    float*    logits  = (float*)take((size_t)B * KEXP * 4);
    float*    gate    = (float*)take((size_t)B * KEXP * 4);
    float*    eo      = (float*)take((size_t)B * KEXP * 4);
    const int frag1   = KT1 * NT1 * FRAG_DW;                  // gate/exp W1 dwords
    const int frag2   = (HG / 32) * (KEXP / 16) * 32 * 8;     // gate_w2 dwords
    uint32_t* gw1p    = (uint32_t*)take((size_t)frag1 * 4);
    uint32_t* gw2p    = (uint32_t*)take((size_t)frag2 * 4);
    uint32_t* ew1p    = (uint32_t*)take((size_t)KEXP * frag1 * 4);
    float*    accum   = (float*)take(17 * 4);
    (void)ws_size;

    hipMemsetAsync(accum, 0, 17 * sizeof(float), stream);

    // ---- gathers + f16 conversion ----
    {
        int total = B * IN_DIM;
        int grid  = (total + 255) / 256;
        gather_convert_f16<<<grid, 256, 0, stream>>>(input_emb_w, x_idx, xf16, B);
        gather_convert_f16<<<grid, 256, 0, stream>>>(sql_emb_w, sql_idx, sql16, B);
    }

    // ---- weight swizzles ----
    swizzle_weights<<<(frag1 + 255) / 256, 256, 0, stream>>>(gate_w1, gw1p, IN_DIM, HG);
    swizzle_weights<<<(frag2 + 255) / 256, 256, 0, stream>>>(gate_w2, gw2p, HG, KEXP);
    for (int k = 0; k < KEXP; ++k) {
        swizzle_weights<<<(frag1 + 255) / 256, 256, 0, stream>>>(
            exp_w1 + (size_t)k * IN_DIM * HE, ew1p + (size_t)k * frag1, IN_DIM, HE);
    }

    // ---- gate MLP (WMMA) ----
    gate_mlp1_wmma<<<B / (16 * 8), 256, 0, stream>>>(sql16, gw1p, gate_b1, h16, B);
    {
        int waves = B / 16;
        int grid  = (waves * 32 + 255) / 256;
        gate_mlp2_wmma<<<grid, 256, 0, stream>>>(h16, gw2p, gate_b2, logits, B);
    }

    // ---- entmax + loss partials ----
    entmax_bisect_kernel<<<(B + 255) / 256, 256, 0, stream>>>(logits, gate, accum, B);

    // ---- fused experts (WMMA, async-LDS staged weights) ----
    expert_fused_wmma<<<(B / (16 * 8)) * KEXP, 256, 0, stream>>>(
        xf16, ew1p, exp_b1, exp_w2, exp_b2, eo, B);

    // ---- combine + losses ----
    finalize_kernel<<<(B + 255) / 256, 256, 0, stream>>>(gate, eo, accum, out, B);
    (void)out_size; (void)n_in;
}
